// CrossCondGPT2_2part_64905545777876
// MI455X (gfx1250) — compile-verified, matrix operand order checked by reading, not run on hardware
//
#include <hip/hip_runtime.h>
#include <math.h>

// ---------------- model constants (from reference) ----------------
#define BATCH   4
#define TSEG    512
#define BLOCKP  512      // pos_emb block stride
#define CH      768
#define HEADS   12
#define HD      64       // CH / HEADS
#define NMC     438      // cond feature dim
#define NMOD    3
#define TT      (NMOD * TSEG)   // 1536
#define LBASE   6
#define VB      512
#define VH      512
#define FFDIM   (4 * CH)        // 3072

typedef _Float16 h16;
typedef __attribute__((ext_vector_type(16))) _Float16 v16h;
typedef __attribute__((ext_vector_type(8)))  _Float16 v8h;
typedef __attribute__((ext_vector_type(4)))  _Float16 v4h;
typedef __attribute__((ext_vector_type(8)))  float    v8f;

// 16-bit A/B WMMA fragment (ISA 7.12.2): v16h element j maps to
// K = hi*8 + j (j<8) and K = 16 + hi*8 + (j-8) (j>=8): two contiguous
// 8-half (16B) chunks per lane -> load as 2x ds_load_b128.
__device__ __forceinline__ v16h frag_ld(const h16* row, int hi) {
  v8h lo = *(const v8h*)(row + (hi << 3));
  v8h hh = *(const v8h*)(row + 16 + (hi << 3));
  return __builtin_shufflevector(lo, hh, 0, 1, 2, 3, 4, 5, 6, 7,
                                 8, 9, 10, 11, 12, 13, 14, 15);
}

// ---------------- embedding: cond@W + gathers + pos ----------------
__global__ void embed_kernel(const int* __restrict__ idx_body,
                             const int* __restrict__ idx_hands,
                             const float* __restrict__ cond,
                             const float* __restrict__ tok_body,
                             const float* __restrict__ tok_hands,
                             const float* __restrict__ pos_emb,
                             const float* __restrict__ cond_W,
                             const float* __restrict__ cond_b,
                             float* __restrict__ x) {
  long i = (long)blockIdx.x * blockDim.x + threadIdx.x;
  if (i >= (long)BATCH * TT * CH) return;
  int c = (int)(i % CH);
  long row = i / CH;
  int t = (int)(row % TT);
  int b = (int)(row / TT);
  int seg = t / TSEG, p = t % TSEG;
  float val;
  if (seg == 0) {
    float s = cond_b[c];
    const float* cr = cond + ((long)b * TSEG + p) * NMC;
    for (int kk = 0; kk < NMC; ++kk) s += cr[kk] * cond_W[(long)kk * CH + c];
    val = s;
  } else if (seg == 1) {
    val = tok_body[(long)idx_body[b * TSEG + p] * CH + c];
  } else {
    val = tok_hands[(long)idx_hands[b * TSEG + p] * CH + c];
  }
  val += pos_emb[(long)(seg * BLOCKP + p) * CH + c];
  x[i] = val;
}

// ---------------- LayerNorm ----------------
__global__ __launch_bounds__(256) void ln_kernel(const float* __restrict__ x,
                                                 const float* __restrict__ w,
                                                 const float* __restrict__ bvec,
                                                 float* __restrict__ out) {
  __shared__ float red[256];
  long row = blockIdx.x;
  const float* xr = x + row * CH;
  float s = 0.f;
  for (int c = threadIdx.x; c < CH; c += 256) s += xr[c];
  red[threadIdx.x] = s; __syncthreads();
  for (int o = 128; o > 0; o >>= 1) {
    if ((int)threadIdx.x < o) red[threadIdx.x] += red[threadIdx.x + o];
    __syncthreads();
  }
  float mu = red[0] / CH;
  __syncthreads();
  float vs = 0.f;
  for (int c = threadIdx.x; c < CH; c += 256) { float d = xr[c] - mu; vs += d * d; }
  red[threadIdx.x] = vs; __syncthreads();
  for (int o = 128; o > 0; o >>= 1) {
    if ((int)threadIdx.x < o) red[threadIdx.x] += red[threadIdx.x + o];
    __syncthreads();
  }
  float rstd = rsqrtf(red[0] / CH + 1e-5f);
  for (int c = threadIdx.x; c < CH; c += 256)
    out[row * CH + c] = (xr[c] - mu) * rstd * w[c] + bvec[c];
}

// ---------------- WMMA GEMM: out = act(A@W + bias) (+ residual) ----------------
// A: M x K fp32, W: K x N fp32 (both row major). fp32 streamed from HBM once,
// converted to f16 into LDS; W tile stored TRANSPOSED so all fragment loads
// are contiguous 16B chunks (ds_load_b128). 64x64x32 tiles, 4 waves/block.
#define BM 64
#define BN 64
#define BKK 32
#define LDA 40   // halves; 80B row stride (16B multiple)
#define LDBT 40

__global__ __launch_bounds__(128) void gemm_kernel(
    const float* __restrict__ A, const float* __restrict__ W,
    const float* __restrict__ bias, const float* __restrict__ residual,
    float* __restrict__ out, int M, int N, int K, int act) {
  __shared__ __align__(16) h16 As[BM][LDA];     // [m][k]
  __shared__ __align__(16) h16 BsT[BN][LDBT];   // [n][k] (transposed)
  const int tid = threadIdx.x;
  const int lane = tid & 31, wid = tid >> 5;
  const int m15 = lane & 15, hi = lane >> 4;
  const int tileM = blockIdx.y * BM, tileN = blockIdx.x * BN;
  const int waveM = (wid >> 1) * 32, waveN = (wid & 1) * 32;
  v8f acc00 = {}, acc01 = {}, acc10 = {}, acc11 = {};
  const int ktEnd = (K + BKK - 1) / BKK;
  for (int kt = 0; kt < ktEnd; ++kt) {
    if (kt + 1 < ktEnd) {  // lowers to global_prefetch_b8
      __builtin_prefetch(&A[(long)(tileM + (tid >> 1)) * K + (kt + 1) * BKK], 0, 0);
      __builtin_prefetch(&W[(long)((kt + 1) * BKK + (tid & 31)) * N + tileN], 0, 0);
    }
    // A tile: float4 coalesced loads, packed ds_store_b64
    for (int i = tid * 4; i < BM * BKK; i += 512) {
      int r = i >> 5, c = i & 31;
      int gm = tileM + r, gk = kt * BKK + c;
      v4h pk;
      if (gm < M && gk + 3 < K) {
        float4 f = *(const float4*)&A[(long)gm * K + gk];
        pk = (v4h){(h16)f.x, (h16)f.y, (h16)f.z, (h16)f.w};
      } else {
        for (int e = 0; e < 4; ++e)
          pk[e] = (h16)((gm < M && gk + e < K) ? A[(long)gm * K + gk + e] : 0.f);
      }
      *(v4h*)&As[r][c] = pk;
    }
    // W tile: float4 coalesced loads, transposed scatter into LDS
    for (int i = tid * 4; i < BKK * BN; i += 512) {
      int r = i >> 6, c = i & 63;
      int gk = kt * BKK + r, gn = tileN + c;
      if (gk < K && gn + 3 < N) {
        float4 f = *(const float4*)&W[(long)gk * N + gn];
        BsT[c + 0][r] = (h16)f.x;
        BsT[c + 1][r] = (h16)f.y;
        BsT[c + 2][r] = (h16)f.z;
        BsT[c + 3][r] = (h16)f.w;
      } else {
        for (int e = 0; e < 4; ++e)
          BsT[c + e][r] = (h16)((gk < K && gn + e < N) ? W[(long)gk * N + gn + e] : 0.f);
      }
    }
    __syncthreads();
    v16h a0 = frag_ld(&As[waveM + m15][0], hi);
    v16h a1 = frag_ld(&As[waveM + 16 + m15][0], hi);
    v16h b0 = frag_ld(&BsT[waveN + m15][0], hi);
    v16h b1 = frag_ld(&BsT[waveN + 16 + m15][0], hi);
    acc00 = __builtin_amdgcn_wmma_f32_16x16x32_f16(false, a0, false, b0, (short)0, acc00, false, false);
    acc01 = __builtin_amdgcn_wmma_f32_16x16x32_f16(false, a0, false, b1, (short)0, acc01, false, false);
    acc10 = __builtin_amdgcn_wmma_f32_16x16x32_f16(false, a1, false, b0, (short)0, acc10, false, false);
    acc11 = __builtin_amdgcn_wmma_f32_16x16x32_f16(false, a1, false, b1, (short)0, acc11, false, false);
    __syncthreads();
  }
  v8f accs[2][2] = {acc00, acc01, acc10, acc11};
#pragma unroll
  for (int i = 0; i < 2; ++i)
#pragma unroll
    for (int j = 0; j < 2; ++j)
#pragma unroll
      for (int r = 0; r < 8; ++r) {
        int m = tileM + waveM + i * 16 + r + hi * 8;
        int n = tileN + waveN + j * 16 + m15;
        if (m < M && n < N) {
          float v = accs[i][j][r];
          if (bias) v += bias[n];
          if (act == 1) v = 0.5f * v * (1.f + erff(v * 0.70710678118f));
          if (residual) v += residual[(long)m * N + n];
          out[(long)m * N + n] = v;
        }
      }
}

// ---------------- flash-style WMMA attention ----------------
// 4 waves/block share one K/V LDS stage per 32-key tile; each wave owns a
// 16-query tile: S = Q*K^T (two k=32 WMMAs), online softmax with 16-lane
// shfl reductions, P repacked through per-wave LDS, O += P*V (4 WMMAs).
__global__ __launch_bounds__(128) void attn_kernel(const float* __restrict__ q,
                                                   const float* __restrict__ k,
                                                   const float* __restrict__ v,
                                                   float* __restrict__ y) {
  __shared__ __align__(16) h16 Kls[32][72];      // [key][d]   (144B rows)
  __shared__ __align__(16) h16 Vls[64][40];      // [d][key]   (transposed)
  __shared__ __align__(16) h16 Pt[4][16][40];    // per-wave P scratch
  const int tid = threadIdx.x;
  const int lane = tid & 31, wid = tid >> 5;
  const int m15 = lane & 15, hi = lane >> 4;
  const int b = blockIdx.y / HEADS, h = blockIdx.y % HEADS;
  const int qt = blockIdx.x * 4 + wid;           // 16-query tile of this wave
  const int lqBase = (qt * 16) % TSEG;

  // Q fragments (pre-scaled by 1/sqrt(64)); contiguous float chunks.
  const float* qrow = q + ((long)b * TT + qt * 16 + m15) * CH + h * HD;
  v16h aQ0, aQ1;
#pragma unroll
  for (int j = 0; j < 8; ++j) {
    aQ0[j]     = (h16)(qrow[hi * 8 + j] * 0.125f);
    aQ0[j + 8] = (h16)(qrow[16 + hi * 8 + j] * 0.125f);
    aQ1[j]     = (h16)(qrow[32 + hi * 8 + j] * 0.125f);
    aQ1[j + 8] = (h16)(qrow[48 + hi * 8 + j] * 0.125f);
  }

  float Mrow[8], Lrow[8], sc[8];
  v8f O0 = {}, O1 = {}, O2 = {}, O3 = {};
#pragma unroll
  for (int r = 0; r < 8; ++r) { Mrow[r] = -INFINITY; Lrow[r] = 0.f; }

  for (int jt = 0; jt < TT / 32; ++jt) {
    // ---- cooperative K/V staging (coalesced float4 loads) ----
    const long kvBase = ((long)b * TT + jt * 32) * CH + h * HD;
    for (int i = tid * 4; i < 32 * 64; i += 512) {
      int kk = i >> 6, d = i & 63;
      float4 f = *(const float4*)&k[kvBase + (long)kk * CH + d];
      *(v4h*)&Kls[kk][d] = (v4h){(h16)f.x, (h16)f.y, (h16)f.z, (h16)f.w};
    }
    for (int i = tid * 4; i < 32 * 64; i += 512) {
      int kk = i >> 6, d = i & 63;
      float4 f = *(const float4*)&v[kvBase + (long)kk * CH + d];
      Vls[d + 0][kk] = (h16)f.x;
      Vls[d + 1][kk] = (h16)f.y;
      Vls[d + 2][kk] = (h16)f.z;
      Vls[d + 3][kk] = (h16)f.w;
    }
    __syncthreads();

    // ---- S = Q*K^T for the two 16-key halves ----
    v8f S[2];
#pragma unroll
    for (int s2 = 0; s2 < 2; ++s2) {
      v16h bK0 = frag_ld(&Kls[s2 * 16 + m15][0], hi);
      v16h bK1 = frag_ld(&Kls[s2 * 16 + m15][32], hi);
      v8f c = {};
      c = __builtin_amdgcn_wmma_f32_16x16x32_f16(false, aQ0, false, bK0, (short)0, c, false, false);
      c = __builtin_amdgcn_wmma_f32_16x16x32_f16(false, aQ1, false, bK1, (short)0, c, false, false);
      const int lk = ((jt * 32 + s2 * 16) % TSEG) + m15;  // tiled causal mask
#pragma unroll
      for (int r = 0; r < 8; ++r) {
        int lqr = lqBase + r + 8 * hi;
        if (lk > lqr) c[r] = -INFINITY;
      }
      S[s2] = c;
    }

    // ---- online softmax (16-lane row reductions) ----
#pragma unroll
    for (int r = 0; r < 8; ++r) {
      float mx = fmaxf(S[0][r], S[1][r]);
#pragma unroll
      for (int off = 1; off < 16; off <<= 1) mx = fmaxf(mx, __shfl_xor(mx, off, 32));
      float newM = fmaxf(Mrow[r], mx);
      float scale = __expf(Mrow[r] - newM);
      float p0 = __expf(S[0][r] - newM);
      float p1 = __expf(S[1][r] - newM);
      float ps = p0 + p1;
#pragma unroll
      for (int off = 1; off < 16; off <<= 1) ps += __shfl_xor(ps, off, 32);
      Mrow[r] = newM;
      Lrow[r] = Lrow[r] * scale + ps;
      sc[r] = scale;
      Pt[wid][r + 8 * hi][m15] = (h16)p0;
      Pt[wid][r + 8 * hi][16 + m15] = (h16)p1;
    }
#pragma unroll
    for (int r = 0; r < 8; ++r) { O0[r] *= sc[r]; O1[r] *= sc[r]; O2[r] *= sc[r]; O3[r] *= sc[r]; }

    // ---- O += P * V ---- (same-wave LDS RAW: DS ops are in-order per wave)
    v16h aP = frag_ld(&Pt[wid][m15][0], hi);
    v16h bV0 = frag_ld(&Vls[0 + m15][0], hi);
    v16h bV1 = frag_ld(&Vls[16 + m15][0], hi);
    v16h bV2 = frag_ld(&Vls[32 + m15][0], hi);
    v16h bV3 = frag_ld(&Vls[48 + m15][0], hi);
    O0 = __builtin_amdgcn_wmma_f32_16x16x32_f16(false, aP, false, bV0, (short)0, O0, false, false);
    O1 = __builtin_amdgcn_wmma_f32_16x16x32_f16(false, aP, false, bV1, (short)0, O1, false, false);
    O2 = __builtin_amdgcn_wmma_f32_16x16x32_f16(false, aP, false, bV2, (short)0, O2, false, false);
    O3 = __builtin_amdgcn_wmma_f32_16x16x32_f16(false, aP, false, bV3, (short)0, O3, false, false);
    __syncthreads();
  }

  const long baseY = ((long)b * TT + qt * 16) * CH + h * HD;
#pragma unroll
  for (int r = 0; r < 8; ++r) {
    const long rb = baseY + (long)(r + 8 * hi) * CH + m15;
    float invL = 1.f / Lrow[r];
    y[rb]      = O0[r] * invL;
    y[rb + 16] = O1[r] * invL;
    y[rb + 32] = O2[r] * invL;
    y[rb + 48] = O3[r] * invL;
  }
}

// ---------------- pack modality rows for the output heads ----------------
__global__ void pack_kernel(const float* __restrict__ src, float* __restrict__ dst, int seg) {
  long i = (long)blockIdx.x * blockDim.x + threadIdx.x;
  if (i >= (long)BATCH * TSEG * CH) return;
  int c = (int)(i % CH);
  long row = i / CH;
  int t = (int)(row % TSEG);
  int b = (int)(row / TSEG);
  dst[i] = src[((long)b * TT + seg * TSEG + t) * CH + c];
}

// ---------------- driver ----------------
extern "C" void kernel_launch(void* const* d_in, const int* in_sizes, int n_in,
                              void* d_out, int out_size, void* d_ws, size_t ws_size,
                              hipStream_t stream) {
  (void)in_sizes; (void)n_in; (void)out_size; (void)ws_size;
  const int*   idx_body  = (const int*)d_in[0];
  const int*   idx_hands = (const int*)d_in[1];
  const float* cond      = (const float*)d_in[2];
  const float* tok_body  = (const float*)d_in[3];
  const float* tok_hands = (const float*)d_in[4];
  const float* pos_emb   = (const float*)d_in[5];
  const float* cond_W    = (const float*)d_in[6];
  const float* cond_b    = (const float*)d_in[7];
  // base_params leaves: d_in[8..23], head_params leaves: d_in[24..39]
  const float* lnf_w      = (const float*)d_in[40];
  const float* lnf_b      = (const float*)d_in[41];
  const float* Wout_body  = (const float*)d_in[42];
  const float* Wout_hands = (const float*)d_in[43];
  float* outp = (float*)d_out;

  const long M  = (long)BATCH * TT;       // 6144
  const long MC = M * CH;
  float* ws = (float*)d_ws;
  float* X  = ws;
  float* Hb = ws + 1 * MC;   // LN output; reused as attention output Y
  float* Q  = ws + 2 * MC;   // reused as packed body rows
  float* Kb = ws + 3 * MC;   // reused as packed hands rows
  float* Vb = ws + 4 * MC;
  float* FF = ws + 5 * MC;   // 4*MC floats
  float* Y  = Hb;

  embed_kernel<<<(unsigned)((MC + 255) / 256), 256, 0, stream>>>(
      idx_body, idx_hands, cond, tok_body, tok_hands, pos_emb, cond_W, cond_b, X);

  dim3 gC((CH + BN - 1) / BN, (unsigned)((M + BM - 1) / BM));      // 12 x 96
  dim3 gF((FFDIM + BN - 1) / BN, (unsigned)((M + BM - 1) / BM));   // 48 x 96
  dim3 gA(TT / 64, BATCH * HEADS);                                 // 24 x 48 (4 q-tiles/block)

  for (int li = 0; li < LBASE + LBASE; ++li) {
    const int base = (li < LBASE);
    const int i = base ? li : li - LBASE;
    const int o = base ? 8 : 24;
    const float* ln1w = (const float*)d_in[o + 0]  + (long)i * CH;
    const float* ln1b = (const float*)d_in[o + 1]  + (long)i * CH;
    const float* Wq   = (const float*)d_in[o + 2]  + (long)i * CH * CH;
    const float* bq   = (const float*)d_in[o + 3]  + (long)i * CH;
    const float* Wk   = (const float*)d_in[o + 4]  + (long)i * CH * CH;
    const float* bk   = (const float*)d_in[o + 5]  + (long)i * CH;
    const float* Wv   = (const float*)d_in[o + 6]  + (long)i * CH * CH;
    const float* bv   = (const float*)d_in[o + 7]  + (long)i * CH;
    const float* Wp   = (const float*)d_in[o + 8]  + (long)i * CH * CH;
    const float* bp   = (const float*)d_in[o + 9]  + (long)i * CH;
    const float* ln2w = (const float*)d_in[o + 10] + (long)i * CH;
    const float* ln2b = (const float*)d_in[o + 11] + (long)i * CH;
    const float* W1   = (const float*)d_in[o + 12] + (long)i * CH * FFDIM;
    const float* b1   = (const float*)d_in[o + 13] + (long)i * FFDIM;
    const float* W2   = (const float*)d_in[o + 14] + (long)i * FFDIM * CH;
    const float* b2   = (const float*)d_in[o + 15] + (long)i * CH;

    ln_kernel<<<(unsigned)M, 256, 0, stream>>>(X, ln1w, ln1b, Hb);
    gemm_kernel<<<gC, 128, 0, stream>>>(Hb, Wq, bq, nullptr, Q,  (int)M, CH, CH, 0);
    gemm_kernel<<<gC, 128, 0, stream>>>(Hb, Wk, bk, nullptr, Kb, (int)M, CH, CH, 0);
    gemm_kernel<<<gC, 128, 0, stream>>>(Hb, Wv, bv, nullptr, Vb, (int)M, CH, CH, 0);
    attn_kernel<<<gA, 128, 0, stream>>>(Q, Kb, Vb, Y);
    gemm_kernel<<<gC, 128, 0, stream>>>(Y, Wp, bp, X, X, (int)M, CH, CH, 0);
    ln_kernel<<<(unsigned)M, 256, 0, stream>>>(X, ln2w, ln2b, Hb);
    gemm_kernel<<<gF, 128, 0, stream>>>(Hb, W1, b1, nullptr, FF, (int)M, FFDIM, CH, 1);
    gemm_kernel<<<gC, 128, 0, stream>>>(FF, W2, b2, X, X, (int)M, CH, FFDIM, 0);
  }

  ln_kernel<<<(unsigned)M, 256, 0, stream>>>(X, lnf_w, lnf_b, Hb);

  const long M2 = (long)BATCH * TSEG;     // 2048
  pack_kernel<<<(unsigned)((M2 * CH + 255) / 256), 256, 0, stream>>>(Hb, Q, 1);
  pack_kernel<<<(unsigned)((M2 * CH + 255) / 256), 256, 0, stream>>>(Hb, Kb, 2);
  dim3 gO((VB + BN - 1) / BN, (unsigned)((M2 + BM - 1) / BM));
  gemm_kernel<<<gO, 128, 0, stream>>>(Q,  Wout_body,  nullptr, nullptr, outp,           (int)M2, VB, CH, 0);
  gemm_kernel<<<gO, 128, 0, stream>>>(Kb, Wout_hands, nullptr, nullptr, outp + M2 * VB, (int)M2, VH, CH, 0);
}
